// PINN_LORA_77146202571399
// MI455X (gfx1250) — matrix operand order repaired and checked by reading, not compiled, not run on hardware
//
#include <hip/hip_runtime.h>
#include <cstddef>
#include <cstdint>

typedef _Float16 v16h __attribute__((ext_vector_type(16)));
typedef _Float16 v8h  __attribute__((ext_vector_type(8)));
typedef float    v8f  __attribute__((ext_vector_type(8)));

// Packed-weight offsets (in halves) for the 13 matrices of one branch:
// [U0,V0,H0,Z0, U1,V1,H1,Z1, U2,V2,H2,Z2, Wout]
static constexpr int K_OFF[13] = {0, 8192, 16384, 24576, 90112, 155648,
                                  221184, 286720, 352256, 417792, 483328,
                                  548864, 614400};
static constexpr int BR_STRIDE = 712704;   // halves per branch
static constexpr int WAVE_LDS  = 57344;    // bytes of LDS per wave
static constexpr int WG_LDS    = WAVE_LDS * 4;

struct KParams {
  const float* inp;
  float* out;
  const _Float16* pack;
  const float* bias[39];   // [branch*13 + matrix]
};

// ---------------- device helpers ----------------

// B fragment: uniform base + 32-bit per-lane byte offset (SADDR+voffset form).
static __device__ __forceinline__ v16h load_b_frag(const _Float16* __restrict__ w,
                                                   int idx) {
  union { v16h v; uint4 q[2]; } u;
  const int off = idx << 5;   // 32 bytes per fragment-lane
  const char* pc = reinterpret_cast<const char*>(w) + off;
  u.q[0] = *reinterpret_cast<const uint4*>(pc);
  u.q[1] = *reinterpret_cast<const uint4*>(pc + 16);
  return u.v;
}

// A-fragment (16x32 f16) per ISA layout: lane m = L&15, hs = L>>4,
// element e -> K = kt*32 + 8*hs + e (+8 for e>=8)  => two 16B LDS loads.
static __device__ __forceinline__ v16h load_a_frag(const _Float16* __restrict__ row,
                                                   int kt, int hs) {
  union { v16h v; uint4 q[2]; } u;
  const char* pc = reinterpret_cast<const char*>(row) + kt * 64 + hs * 16;
  u.q[0] = *reinterpret_cast<const uint4*>(pc);
  u.q[1] = *reinterpret_cast<const uint4*>(pc + 32);
  return u.v;
}

static __device__ __forceinline__ v8f wmma_f16(v16h a, v16h b, v8f c) {
  return __builtin_amdgcn_wmma_f32_16x16x32_f16(false, a, false, b, (short)0, c,
                                                false, false);
}

static __device__ __forceinline__ float tanh_fast(float x) {
  float e = __expf(2.0f * x);
  return 1.0f - 2.0f * __builtin_amdgcn_rcpf(e + 1.0f);
}

// One GEMM over NT 16x16 output tiles; 4 independent accumulator chains per
// step to cover WMMA result-hazard windows and deepen the B-load pipeline.
template <int KP, int NT, class F>
static __device__ __forceinline__ void gemm_tiles(const v16h (&a)[KP],
                                                  const _Float16* __restrict__ w,
                                                  const float* __restrict__ bias,
                                                  int ln, int L, F&& post) {
  for (int nt = 0; nt < NT; nt += 4) {
    if (nt + 4 < NT)  // prefetch next group of B tiles (L2 weight stream)
      __builtin_prefetch(w + (size_t)(nt + 4) * KP * 512, 0, 0);
    v8f acc[4];
#pragma unroll
    for (int t = 0; t < 4; ++t) {
      float b = bias[(nt + t) * 16 + ln];
#pragma unroll
      for (int e = 0; e < 8; ++e) acc[t][e] = b;
    }
#pragma unroll
    for (int kt = 0; kt < KP; ++kt) {
#pragma unroll
      for (int t = 0; t < 4; ++t)
        acc[t] =
            wmma_f16(a[kt], load_b_frag(w, ((nt + t) * KP + kt) * 32 + L), acc[t]);
    }
#pragma unroll
    for (int t = 0; t < 4; ++t) post(nt + t, acc[t]);
  }
}

// One modified-MLP block: U,V,H from h0 (KP k-tiles), Z from tanh(H), then
// h0 <- (1-Z)*U + Z*V.  All LDS tiles are per-wave.
template <int KP>
static __device__ __forceinline__ void run_block(
    _Float16* h0, _Float16* h1, _Float16* Ub, _Float16* Vb,
    const _Float16* wU, const _Float16* wV, const _Float16* wH,
    const _Float16* wZ, const float* bU, const float* bV, const float* bH,
    const float* bZ, int ln, int hs, int L) {
  v16h a[KP];
#pragma unroll
  for (int kt = 0; kt < KP; ++kt) a[kt] = load_a_frag(h0 + ln * 256, kt, hs);

  gemm_tiles<KP, 16>(a, wU, bU, ln, L, [&](int nt, v8f acc) {
    union { v8h v; uint4 q; } o;
#pragma unroll
    for (int e = 0; e < 8; ++e) o.v[e] = (_Float16)tanh_fast(acc[e]);
    reinterpret_cast<uint4*>(Ub)[nt * 32 + L] = o.q;
  });
  gemm_tiles<KP, 16>(a, wV, bV, ln, L, [&](int nt, v8f acc) {
    union { v8h v; uint4 q; } o;
#pragma unroll
    for (int e = 0; e < 8; ++e) o.v[e] = (_Float16)tanh_fast(acc[e]);
    reinterpret_cast<uint4*>(Vb)[nt * 32 + L] = o.q;
  });
  gemm_tiles<KP, 16>(a, wH, bH, ln, L, [&](int nt, v8f acc) {
#pragma unroll
    for (int e = 0; e < 8; ++e)
      h1[(e + 8 * hs) * 256 + nt * 16 + ln] = (_Float16)tanh_fast(acc[e]);
  });
  __syncthreads();

  v16h ah[8];
#pragma unroll
  for (int kt = 0; kt < 8; ++kt) ah[kt] = load_a_frag(h1 + ln * 256, kt, hs);
  gemm_tiles<8, 16>(ah, wZ, bZ, ln, L, [&](int nt, v8f acc) {
    union { v8h v; uint4 q; } uu, vv;
    uu.q = reinterpret_cast<const uint4*>(Ub)[nt * 32 + L];
    vv.q = reinterpret_cast<const uint4*>(Vb)[nt * 32 + L];
#pragma unroll
    for (int e = 0; e < 8; ++e) {
      float z  = tanh_fast(acc[e]);
      float hn = (1.0f - z) * (float)uu.v[e] + z * (float)vv.v[e];
      h0[(e + 8 * hs) * 256 + nt * 16 + ln] = (_Float16)hn;
    }
  });
  __syncthreads();
}

// ---------------- kernels ----------------

// Pack f32 weight matrix (d x h, row major) into WMMA B-fragment order f16:
// dst[((nt*Kp + kt)*32 + lane)*16 + e] = W[kt*32 + e + 8*(lane>>4) + (e>=8 ? 8 : 0)][nt*16 + lane&15]
__global__ void pack_weights(const float* __restrict__ src,
                             _Float16* __restrict__ dst, int d, int h, int Kp) {
  const int frag = blockIdx.x;
  const int kt = frag % Kp;
  const int nt = frag / Kp;
  const int L = threadIdx.x;
  const int n = nt * 16 + (L & 15);
  const int kbase = kt * 32 + ((L >> 4) << 3);
  _Float16* out = dst + (size_t)(frag * 32 + L) * 16;
#pragma unroll
  for (int e = 0; e < 16; ++e) {
    const int k = kbase + e + ((e >> 3) << 3);
    float v = (k < d) ? src[(size_t)k * h + n] : 0.0f;
    out[e] = (_Float16)v;
  }
}

__global__ __launch_bounds__(128, 1) void pinn_lora_fused(KParams p) {
  extern __shared__ char smem[];
  const int tid = threadIdx.x;
  const int wv  = tid >> 5;
  const int L   = tid & 31;
  const int ln  = L & 15;   // output column within 16x16 tile
  const int hs  = L >> 4;   // half-select

  char* wbase = smem + wv * WAVE_LDS;
  _Float16* h0 = reinterpret_cast<_Float16*>(wbase);          // 16x256 activations
  _Float16* h1 = reinterpret_cast<_Float16*>(wbase + 8192);   // 16x256 tanh(H)
  _Float16* Ub = reinterpret_cast<_Float16*>(wbase + 16384);  // D-frag order U
  _Float16* Vb = reinterpret_cast<_Float16*>(wbase + 24576);  // D-frag order V
  float*    P  = reinterpret_cast<float*>(wbase + 32768);     // 16x384 product

  const int rowBase = blockIdx.x * 64 + wv * 16;

  for (int br = 0; br < 3; ++br) {
    // ---- build input features (K padded to 32, zero beyond in_dim) ----
    if (L < 16) {
      const int row = rowBase + L;
      float f[32];
#pragma unroll
      for (int j = 0; j < 32; ++j) f[j] = 0.0f;
      if (br == 0) {
        f[0] = p.inp[row * 3 + 0];
      } else {
        float v = p.inp[row * 3 + br];
        f[0] = 1.0f;
#pragma unroll
        for (int j = 1; j <= 10; ++j) {
          f[j]      = __sinf(v * (float)j);
          f[10 + j] = __cosf(v * (float)j);
        }
      }
#pragma unroll
      for (int k = 0; k < 32; ++k) h0[L * 256 + k] = (_Float16)f[k];
    }
    __syncthreads();

    const _Float16* wbr = p.pack + (size_t)br * BR_STRIDE;
    const int bb = br * 13;

    // ---- block 0: input K padded to one 32-wide tile ----
    run_block<1>(h0, h1, Ub, Vb, wbr + K_OFF[0], wbr + K_OFF[1], wbr + K_OFF[2],
                 wbr + K_OFF[3], p.bias[bb + 0], p.bias[bb + 1], p.bias[bb + 2],
                 p.bias[bb + 3], ln, hs, L);
    // ---- blocks 1,2: full 256-wide K ----
#pragma unroll
    for (int s = 1; s < 3; ++s)
      run_block<8>(h0, h1, Ub, Vb, wbr + K_OFF[4 * s + 0], wbr + K_OFF[4 * s + 1],
                   wbr + K_OFF[4 * s + 2], wbr + K_OFF[4 * s + 3],
                   p.bias[bb + 4 * s + 0], p.bias[bb + 4 * s + 1],
                   p.bias[bb + 4 * s + 2], p.bias[bb + 4 * s + 3], ln, hs, L);

    // ---- output layer 256 -> 384 (linear), combine into product buffer ----
    {
      v16h a[8];
#pragma unroll
      for (int kt = 0; kt < 8; ++kt) a[kt] = load_a_frag(h0 + ln * 256, kt, hs);
      gemm_tiles<8, 24>(a, wbr + K_OFF[12], p.bias[bb + 12], ln, L,
                        [&](int nt, v8f acc) {
#pragma unroll
                          for (int e = 0; e < 8; ++e) {
                            float* pp = &P[(e + 8 * hs) * 384 + nt * 16 + ln];
                            if (br == 0) *pp = acc[e];
                            else         *pp = *pp * acc[e];
                          }
                        });
      __syncthreads();
    }
  }

  // ---- rank-128 reduction: pred[m, o] = sum_r P[m][o*128 + r] ----
  for (int t = L; t < 48; t += 32) {
    const int m = t / 3;
    const int o = t - m * 3;
    const float* pr = P + m * 384 + o * 128;
    float s = 0.0f;
#pragma unroll 8
    for (int r = 0; r < 128; ++r) s += pr[r];
    p.out[(size_t)(rowBase + m) * 3 + o] = s;
  }
}

// ---------------- host ----------------

extern "C" void kernel_launch(void* const* d_in, const int* in_sizes, int n_in,
                              void* d_out, int out_size, void* d_ws, size_t ws_size,
                              hipStream_t stream) {
  (void)in_sizes; (void)n_in; (void)out_size; (void)ws_size;
  _Float16* pack = reinterpret_cast<_Float16*>(d_ws);

  KParams p;
  p.inp  = reinterpret_cast<const float*>(d_in[0]);
  p.out  = reinterpret_cast<float*>(d_out);
  p.pack = pack;

  // d_in pytree order: inp, then per branch: 3 blocks x (Uw,Ub,Vw,Vb,Hw,Hb,Zw,Zb),
  // Wout, bout  => 26 arrays per branch.
  for (int br = 0; br < 3; ++br) {
    const int base = 1 + br * 26;
    const int din  = (br == 0) ? 1 : 21;
    for (int m = 0; m < 13; ++m) {
      int srcW, srcB, d, h, Kp;
      if (m < 12) {
        const int blk = m / 4, j = m % 4;
        srcW = base + blk * 8 + j * 2;
        srcB = srcW + 1;
        d  = (blk == 0 && j < 3) ? din : 256;
        h  = 256;
        Kp = (blk == 0 && j < 3) ? 1 : 8;
      } else {
        srcW = base + 24; srcB = base + 25;
        d = 256; h = 384; Kp = 8;
      }
      p.bias[br * 13 + m] = reinterpret_cast<const float*>(d_in[srcB]);
      _Float16* dst = pack + (size_t)br * BR_STRIDE + K_OFF[m];
      const int frags = (h / 16) * Kp;
      hipLaunchKernelGGL(pack_weights, dim3(frags), dim3(32), 0, stream,
                         reinterpret_cast<const float*>(d_in[srcW]), dst, d, h, Kp);
    }
  }

  hipFuncSetAttribute(reinterpret_cast<const void*>(pinn_lora_fused),
                      hipFuncAttributeMaxDynamicSharedMemorySize, WG_LDS);
  hipLaunchKernelGGL(pinn_lora_fused, dim3(65536 / 64), dim3(128), WG_LDS, stream, p);
}
